// CNN_BiLSTM_AttenQ_37288906064022
// MI455X (gfx1250) — matrix operand order repaired and checked by reading, not compiled, not run on hardware
//
#include <hip/hip_runtime.h>
#include <hip/hip_bf16.h>

// ---------------------------------------------------------------------------
// CDNA5 (gfx1250) implementation: all matmul-shaped work runs through
// v_wmma_f32_16x16x32_bf16 (wave32 WMMA). Weights pre-packed fragment-major
// so every operand fragment loads as b128s; activations bf16 row-major
// [pos*128+batch][C]. GEMM tile counts + epilogue are template params.
// Recurrent LSTM: persistent per-batch-block workgroups, U held entirely in
// VGPRs, h exchanged via LDS, gate pre-activations prefetched one step ahead.
// ---------------------------------------------------------------------------

typedef __bf16 bf16;
typedef __attribute__((ext_vector_type(16))) __bf16 v16bf;
typedef __attribute__((ext_vector_type(8)))  float  v8f;

union Frag16 { v16bf v; unsigned int w[8]; uint4 q[2]; };
union AccU   { v8f v; float4 q[2]; float f[8]; };

__device__ __forceinline__ float hsig(float z) {
  // piecewise-linear hard sigmoid: clip(0.25*z + 0.5, 0, 1)
  return fminf(fmaxf(0.25f * z + 0.5f, 0.0f), 1.0f);
}

// ---------------------------------------------------------------------------
// Fragment-major packed weight layout (per tap, K multiple of 32):
//   Wp[ ((t*(K/32) + (k>>5))*N + n)*32 + (k&31) ] = W_t[k][n]
// A lane (n = lane&15, kh = lane>>4) of the WMMA B-operand then reads two
// 16-byte runs at element offsets ((chunk*N + n)*32 + kh*8) and +16 -> 2x b128.
// ---------------------------------------------------------------------------

// Epilogue modes for tap_gemm:
//   MODE 0: f32 store, row-major C[(p*128+b)*N + n]
//   MODE 1: f32 store, transposed C[(p*N + n)*128 + b]  (float4s, LSTM gates)
//   MODE 2: bf16 store, row-major (feeds next GEMM's A operand)
//   MODE 3: bf16 store, row-major, fused BN(scale,bias)+ReLU
template <int NTILES, int MODE>
__global__ __launch_bounds__(256) void tap_gemm_kernel(
    const bf16* __restrict__ A, const bf16* __restrict__ Wp,
    const float* __restrict__ bias, void* __restrict__ Cout,
    const float* __restrict__ bng, const float* __restrict__ bnb,
    int Lin, int K, int N, int shift0, int dshift, int ss, int ntaps)
{
  const int p    = blockIdx.x;
  const int wv   = threadIdx.x >> 5;
  const int lane = threadIdx.x & 31;
  const int mr   = lane & 15;
  const int kh   = lane >> 4;
  const int col0 = blockIdx.y * 128;

  AccU acc[NTILES];
#pragma unroll
  for (int nt = 0; nt < NTILES; ++nt) {
    const float bv = bias ? bias[col0 + nt * 16 + mr] : 0.0f;
#pragma unroll
    for (int j = 0; j < 8; ++j) acc[nt].v[j] = bv;
  }

  const int kchunks = K >> 5;
  for (int t = 0; t < ntaps; ++t) {
    const int ps = p + shift0 + t * dshift;
    const int q  = ss ? (ps >> 1) : ps;
    const bool ok = (!ss || ((ps & 1) == 0)) && (ps >= 0) && (q < Lin);
    if (!ok) continue;

    const bf16* Arow = A + (size_t)(q * 128 + wv * 16 + mr) * K + kh * 8;
    const bf16* Wt   = Wp + (size_t)t * K * N;

    for (int kc = 0; kc < kchunks; ++kc) {
      // A fragment (16-bit A-matrix 16x32 layout): two 16B runs per lane
      Frag16 af;
      af.q[0] = *(const uint4*)(Arow + kc * 32);
      af.q[1] = *(const uint4*)(Arow + kc * 32 + 16);

      Frag16 bq[NTILES];
#pragma unroll
      for (int nt = 0; nt < NTILES; ++nt) {
        const bf16* bp = Wt + ((size_t)kc * N + col0 + nt * 16 + mr) * 32 + kh * 8;
        bq[nt].q[0] = *(const uint4*)(bp);
        bq[nt].q[1] = *(const uint4*)(bp + 16);
      }
#pragma unroll
      for (int nt = 0; nt < NTILES; ++nt)
        acc[nt].v = __builtin_amdgcn_wmma_f32_16x16x32_bf16(
            false, af.v, false, bq[nt].v, (short)0, acc[nt].v, false, false);
    }
  }

  if (MODE == 1) {
    float* Cf = (float*)Cout;
#pragma unroll
    for (int nt = 0; nt < NTILES; ++nt) {
      float* cp = Cf + ((size_t)p * N + col0 + nt * 16 + mr) * 128 + wv * 16 + kh * 8;
      *(float4*)(cp)     = acc[nt].q[0];
      *(float4*)(cp + 4) = acc[nt].q[1];
    }
  } else if (MODE == 0) {
    float* Cf = (float*)Cout;
#pragma unroll
    for (int nt = 0; nt < NTILES; ++nt) {
#pragma unroll
      for (int j = 0; j < 8; ++j)
        Cf[(size_t)(p * 128 + wv * 16 + kh * 8 + j) * N + col0 + nt * 16 + mr] =
            acc[nt].v[j];
    }
  } else {
    bf16* Cb = (bf16*)Cout;
#pragma unroll
    for (int nt = 0; nt < NTILES; ++nt) {
      const int col = col0 + nt * 16 + mr;
      float sc = 1.0f, bi = 0.0f;
      if (MODE == 3) { sc = bng[col] * rsqrtf(1.0f + 1e-5f); bi = bnb[col]; }
#pragma unroll
      for (int j = 0; j < 8; ++j) {
        float v = acc[nt].v[j];
        if (MODE == 3) v = fmaxf(v * sc + bi, 0.0f);
        Cb[(size_t)(p * 128 + wv * 16 + kh * 8 + j) * N + col] = (bf16)v;
      }
    }
  }
}

// ---------------------------------------------------------------------------
// Persistent recurrent LSTM kernel: one workgroup per 16-batch block, 8 waves.
// Wave w owns hidden columns w*16..w*16+15 (all 4 gates). U fragments for the
// wave's columns (4 kc x 4 gates x 8 VGPRs = 128 VGPRs) are loaded ONCE into
// registers; c stays in VGPRs across all 512 steps; h is exchanged via 4KB
// LDS each step. Gxt = (x@W + b) stored transposed [pos][512 cols][128 batch]
// so gate loads are float4s; next step's gates are prefetched.
// ---------------------------------------------------------------------------
__global__ __launch_bounds__(256, 1) void lstm_seq_kernel(
    const float* __restrict__ Gxt, const bf16* __restrict__ Up,
    float* __restrict__ carry_h, float* __restrict__ carry_c,
    bf16* __restrict__ hs_out, int Lseq, int rev, int col_off)
{
  __shared__ __align__(16) bf16 Hlds[16 * 128];   // [m(batch in block)][k(hidden)]
  const int wv   = threadIdx.x >> 5;
  const int lane = threadIdx.x & 31;
  const int mr   = lane & 15;
  const int kh   = lane >> 4;
  const int bb0  = blockIdx.x * 16;

  // U fragments: loop-invariant, keep in VGPRs for the whole sequence
  Frag16 uf[4][4];
#pragma unroll
  for (int kc = 0; kc < 4; ++kc) {
#pragma unroll
    for (int g = 0; g < 4; ++g) {
      const bf16* bp = Up + ((size_t)kc * 512 + g * 128 + wv * 16 + mr) * 32 + kh * 8;
      uf[kc][g].q[0] = *(const uint4*)(bp);
      uf[kc][g].q[1] = *(const uint4*)(bp + 16);
    }
  }

  for (int i = threadIdx.x; i < 16 * 128; i += 256) {
    const int m = i >> 7, k = i & 127;
    Hlds[i] = (bf16)carry_h[(size_t)(bb0 + m) * 128 + k];
  }
  float creg[8];
#pragma unroll
  for (int j = 0; j < 8; ++j)
    creg[j] = carry_c[(size_t)(bb0 + kh * 8 + j) * 128 + wv * 16 + mr];
  __syncthreads();

  for (int t = 0; t < Lseq; ++t) {
    const int pos = rev ? (Lseq - 1 - t) : t;

    // gate pre-activations, transposed layout -> 2 float4 loads per gate
    AccU acc[4];
#pragma unroll
    for (int g = 0; g < 4; ++g) {
      const float* gp = Gxt + ((size_t)pos * 512 + g * 128 + wv * 16 + mr) * 128 +
                        bb0 + kh * 8;
      acc[g].q[0] = *(const float4*)(gp);
      acc[g].q[1] = *(const float4*)(gp + 4);
    }
    // prefetch next step's gates into near caches
    if (t + 1 < Lseq) {
      const int pos2 = rev ? (pos - 1) : (pos + 1);
#pragma unroll
      for (int g = 0; g < 4; ++g)
        __builtin_prefetch(Gxt + ((size_t)pos2 * 512 + g * 128 + wv * 16 + mr) * 128 +
                           bb0 + kh * 8, 0, 0);
    }

#pragma unroll
    for (int kc = 0; kc < 4; ++kc) {
      Frag16 af;
      const bf16* hp = Hlds + mr * 128 + kc * 32 + kh * 8;
      af.q[0] = *(const uint4*)(hp);
      af.q[1] = *(const uint4*)(hp + 16);
#pragma unroll
      for (int g = 0; g < 4; ++g)
        acc[g].v = __builtin_amdgcn_wmma_f32_16x16x32_bf16(
            false, af.v, false, uf[kc][g].v, (short)0, acc[g].v, false, false);
    }
    __syncthreads();   // all waves done reading Hlds for this step

#pragma unroll
    for (int j = 0; j < 8; ++j) {
      const float iv = hsig(acc[0].v[j]);
      const float fv = hsig(acc[1].v[j]);
      const float gv = fmaxf(acc[2].v[j], 0.0f);
      const float ov = hsig(acc[3].v[j]);
      const float cv = fv * creg[j] + iv * gv;
      creg[j] = cv;
      const bf16 hb = (bf16)(ov * hsig(cv));
      Hlds[(kh * 8 + j) * 128 + wv * 16 + mr] = hb;
      hs_out[((size_t)pos * 128 + bb0 + kh * 8 + j) * 256 +
             col_off + wv * 16 + mr] = hb;
    }
    __syncthreads();   // h published before next step reads it
  }

#pragma unroll
  for (int j = 0; j < 8; ++j)
    carry_c[(size_t)(bb0 + kh * 8 + j) * 128 + wv * 16 + mr] = creg[j];
  for (int i = threadIdx.x; i < 16 * 128; i += 256) {
    const int m = i >> 7, k = i & 127;
    carry_h[(size_t)(bb0 + m) * 128 + k] = (float)Hlds[i];
  }
}

// ---------------------------------------------------------------------------
// Encoder conv1d(1->128, k=8, s=4, p=3) + BN + ReLU -> bf16 [p][b][c]
// ---------------------------------------------------------------------------
__global__ __launch_bounds__(256) void encoder_kernel(
    const float* __restrict__ x, const float* __restrict__ w,
    const float* __restrict__ bg, const float* __restrict__ bb,
    bf16* __restrict__ xs)
{
  const int idx = blockIdx.x * 256 + threadIdx.x;   // 512*128*128
  const int c = idx & 127;
  const int b = (idx >> 7) & 127;
  const int p = idx >> 14;
  if (p >= 512) return;
  float acc = 0.0f;
#pragma unroll
  for (int k = 0; k < 8; ++k) {
    const int xi = 4 * p + k - 3;
    const float xv = (xi >= 0 && xi < 2048) ? x[(size_t)b * 2048 + xi] : 0.0f;
    acc += w[c * 8 + k] * xv;
  }
  const float s = bg[c] * rsqrtf(1.0f + 1e-5f);
  xs[idx] = (bf16)fmaxf(acc * s + bb[c], 0.0f);
}

// --------------------------- weight pre-packing ----------------------------
// fragment-major: dst[ ((k>>5)*N + n)*32 + (k&31) ]  (single "tap")
__global__ void pack_rowmajor_kernel(const float* __restrict__ src,
                                     bf16* __restrict__ dst, int K, int N)
{
  const int idx = blockIdx.x * 256 + threadIdx.x;
  if (idx >= K * N) return;
  const int k = idx / N, n = idx - k * N;
  dst[((size_t)(k >> 5) * N + n) * 32 + (k & 31)] = (bf16)src[idx];
}

// conv weights -> per-tap fragment-major, output channels padded to Np
__global__ void pack_conv_kernel(const float* __restrict__ src,
                                 bf16* __restrict__ dst,
                                 int O, int I, int Kc, int Np, int transposed)
{
  const int idx = blockIdx.x * 256 + threadIdx.x;
  if (idx >= Kc * I * Np) return;
  const int t = idx / (I * Np);
  const int r = idx - t * I * Np;
  const int i = r / Np;
  const int o = r - i * Np;
  float v = 0.0f;
  if (o < O)
    v = transposed ? src[((size_t)i * O + o) * Kc + t]
                   : src[((size_t)o * I + i) * Kc + t];
  dst[(size_t)t * I * Np + (((size_t)(i >> 5) * Np + o) * 32 + (i & 31))] = (bf16)v;
}

// ------------------------------- elementwise -------------------------------
__global__ void zero_f32_kernel(float* __restrict__ p, int n)
{ const int i = blockIdx.x * 256 + threadIdx.x; if (i < n) p[i] = 0.0f; }

__global__ void final_out_kernel(const float* __restrict__ acc,
                                 const float* __restrict__ g,
                                 const float* __restrict__ b,
                                 float* __restrict__ out)
{
  const int idx = blockIdx.x * 256 + threadIdx.x;   // 128*4*2048
  if (idx >= 128 * 4 * 2048) return;
  const int p  = idx & 2047;
  const int c  = (idx >> 11) & 3;
  const int bb = idx >> 13;
  const float v = acc[((size_t)p * 128 + bb) * 16 + c];
  const float s = g[c] * rsqrtf(1.0f + 1e-5f);
  out[idx] = fmaxf(v * s + b[c], 0.0f);   // out[b][c][p]
}

// ---------------------------------------------------------------------------
extern "C" void kernel_launch(void* const* d_in, const int* in_sizes, int n_in,
                              void* d_out, int out_size, void* d_ws, size_t ws_size,
                              hipStream_t stream)
{
  (void)in_sizes; (void)n_in; (void)out_size; (void)ws_size;
  const int TPB = 256;
  auto NB = [](long long n) { return (int)((n + 255) / 256); };
  const float* nf = nullptr;

  // ---- workspace layout (bump allocated, 256B aligned) ----
  char* ws = (char*)d_ws;
  size_t off = 0;
  auto take = [&](size_t bytes) -> char* {
    char* p = ws + off; off += (bytes + 255) & ~(size_t)255; return p;
  };
  bf16*  xs0  = (bf16*) take(512ull * 128 * 128 * 2);   // encoder out (bf16)
  bf16*  hs0  = (bf16*) take(512ull * 128 * 256 * 2);   // layer-0 bi output
  bf16*  hs1  = (bf16*) take(512ull * 128 * 256 * 2);   // layer-1 bi output
  float* Gx   = (float*)take(512ull * 128 * 512 * 4);   // gates / d22 f32 scratch
  float* ch   = (float*)take(128 * 128 * 4);            // carry h
  float* cc   = (float*)take(128 * 128 * 4);            // carry c
  bf16*  W0fp = (bf16*) take(128ull * 512 * 2);
  bf16*  U0fp = (bf16*) take(128ull * 512 * 2);
  bf16*  W0rp = (bf16*) take(128ull * 512 * 2);
  bf16*  U0rp = (bf16*) take(128ull * 512 * 2);
  bf16*  W1fp = (bf16*) take(256ull * 512 * 2);
  bf16*  U1fp = (bf16*) take(128ull * 512 * 2);
  bf16*  W1rp = (bf16*) take(256ull * 512 * 2);
  bf16*  U1rp = (bf16*) take(128ull * 512 * 2);
  bf16*  up1p = (bf16*) take(8ull * 256 * 128 * 2);
  bf16*  d11p = (bf16*) take(5ull * 128 * 128 * 2);
  bf16*  d12p = (bf16*) take(5ull * 128 * 64 * 2);
  bf16*  up2p = (bf16*) take(8ull * 64 * 32 * 2);
  bf16*  d21p = (bf16*) take(5ull * 32 * 32 * 2);
  bf16*  d22p = (bf16*) take(5ull * 32 * 16 * 2);       // N padded 4 -> 16
  bf16*  Bb0  = (bf16*) take(1024ull * 128 * 128 * 2);  // bf16 ping
  bf16*  Bb1  = (bf16*) take(1024ull * 128 * 128 * 2);  // bf16 pong
  float* F1 = Gx + 1024ull * 128 * 128;                 // d22 f32 out (aliases Gx)

  const float* xin = (const float*)d_in[0];

  // ---- pack weights into fragment-major WMMA B-operand layout ----
  pack_rowmajor_kernel<<<NB(128 * 512), TPB, 0, stream>>>((const float*)d_in[4],  W0fp, 128, 512);
  pack_rowmajor_kernel<<<NB(128 * 512), TPB, 0, stream>>>((const float*)d_in[5],  U0fp, 128, 512);
  pack_rowmajor_kernel<<<NB(128 * 512), TPB, 0, stream>>>((const float*)d_in[7],  W0rp, 128, 512);
  pack_rowmajor_kernel<<<NB(128 * 512), TPB, 0, stream>>>((const float*)d_in[8],  U0rp, 128, 512);
  pack_rowmajor_kernel<<<NB(256 * 512), TPB, 0, stream>>>((const float*)d_in[10], W1fp, 256, 512);
  pack_rowmajor_kernel<<<NB(128 * 512), TPB, 0, stream>>>((const float*)d_in[11], U1fp, 128, 512);
  pack_rowmajor_kernel<<<NB(256 * 512), TPB, 0, stream>>>((const float*)d_in[13], W1rp, 256, 512);
  pack_rowmajor_kernel<<<NB(128 * 512), TPB, 0, stream>>>((const float*)d_in[14], U1rp, 128, 512);
  pack_conv_kernel<<<NB(8 * 256 * 128), TPB, 0, stream>>>((const float*)d_in[16], up1p, 128, 256, 8, 128, 1);
  pack_conv_kernel<<<NB(5 * 128 * 128), TPB, 0, stream>>>((const float*)d_in[17], d11p, 128, 128, 5, 128, 0);
  pack_conv_kernel<<<NB(5 * 128 * 64),  TPB, 0, stream>>>((const float*)d_in[20], d12p,  64, 128, 5,  64, 0);
  pack_conv_kernel<<<NB(8 * 64 * 32),   TPB, 0, stream>>>((const float*)d_in[23], up2p,  32,  64, 8,  32, 1);
  pack_conv_kernel<<<NB(5 * 32 * 32),   TPB, 0, stream>>>((const float*)d_in[24], d21p,  32,  32, 5,  32, 0);
  pack_conv_kernel<<<NB(5 * 32 * 16),   TPB, 0, stream>>>((const float*)d_in[27], d22p,   4,  32, 5,  16, 0);

  // ---- encoder + zero carry ----
  zero_f32_kernel<<<NB(128 * 128), TPB, 0, stream>>>(ch, 128 * 128);
  zero_f32_kernel<<<NB(128 * 128), TPB, 0, stream>>>(cc, 128 * 128);
  encoder_kernel<<<NB(512ll * 128 * 128), TPB, 0, stream>>>(
      xin, (const float*)d_in[1], (const float*)d_in[2], (const float*)d_in[3], xs0);

  // ---- bidirectional LSTM: 4 sequential passes (h/c chained via carry) ----
  // layer 0 forward
  tap_gemm_kernel<8, 1><<<dim3(512, 4), TPB, 0, stream>>>(xs0, W0fp,
      (const float*)d_in[6], Gx, nf, nf, 512, 128, 512, 0, 0, 0, 1);
  lstm_seq_kernel<<<8, TPB, 0, stream>>>(Gx, U0fp, ch, cc, hs0, 512, 0, 0);
  // layer 0 reverse
  tap_gemm_kernel<8, 1><<<dim3(512, 4), TPB, 0, stream>>>(xs0, W0rp,
      (const float*)d_in[9], Gx, nf, nf, 512, 128, 512, 0, 0, 0, 1);
  lstm_seq_kernel<<<8, TPB, 0, stream>>>(Gx, U0rp, ch, cc, hs0, 512, 1, 128);
  // layer 1 forward (input = hs0, K=256)
  tap_gemm_kernel<8, 1><<<dim3(512, 4), TPB, 0, stream>>>(hs0, W1fp,
      (const float*)d_in[12], Gx, nf, nf, 512, 256, 512, 0, 0, 0, 1);
  lstm_seq_kernel<<<8, TPB, 0, stream>>>(Gx, U1fp, ch, cc, hs1, 512, 0, 0);
  // layer 1 reverse
  tap_gemm_kernel<8, 1><<<dim3(512, 4), TPB, 0, stream>>>(hs0, W1rp,
      (const float*)d_in[15], Gx, nf, nf, 512, 256, 512, 0, 0, 0, 1);
  lstm_seq_kernel<<<8, TPB, 0, stream>>>(Gx, U1rp, ch, cc, hs1, 512, 1, 128);

  // ---- decoder: every stage GEMM -> bf16 directly (fused epilogues) ----
  // up1: ConvT(256->128,k8,s2,p3): q=(p+3-k)/2  -> bf16
  tap_gemm_kernel<8, 2><<<dim3(1024, 1), TPB, 0, stream>>>(hs1, up1p, nf, Bb0,
      nf, nf, 512, 256, 128, 3, -1, 1, 8);
  // d11: Conv(128->128,k5,p2) + BN + ReLU -> bf16
  tap_gemm_kernel<8, 3><<<dim3(1024, 1), TPB, 0, stream>>>(Bb0, d11p, nf, Bb1,
      (const float*)d_in[18], (const float*)d_in[19], 1024, 128, 128, -2, 1, 0, 5);
  // d12: Conv(128->64,k5,p2) + BN + ReLU -> bf16
  tap_gemm_kernel<4, 3><<<dim3(1024, 1), TPB, 0, stream>>>(Bb1, d12p, nf, Bb0,
      (const float*)d_in[21], (const float*)d_in[22], 1024, 128, 64, -2, 1, 0, 5);
  // up2: ConvT(64->32,k8,s2,p3) -> bf16
  tap_gemm_kernel<2, 2><<<dim3(2048, 1), TPB, 0, stream>>>(Bb0, up2p, nf, Bb1,
      nf, nf, 1024, 64, 32, 3, -1, 1, 8);
  // d21: Conv(32->32,k5,p2) + BN + ReLU -> bf16
  tap_gemm_kernel<2, 3><<<dim3(2048, 1), TPB, 0, stream>>>(Bb1, d21p, nf, Bb0,
      (const float*)d_in[25], (const float*)d_in[26], 2048, 32, 32, -2, 1, 0, 5);
  // d22: Conv(32->4,k5,p2), N padded to 16 -> f32
  tap_gemm_kernel<1, 0><<<dim3(2048, 1), TPB, 0, stream>>>(Bb0, d22p, nf, F1,
      nf, nf, 2048, 32, 16, -2, 1, 0, 5);
  // BN + ReLU + transpose to [B,4,2048]
  final_out_kernel<<<NB(128ll * 4 * 2048), TPB, 0, stream>>>(F1,
      (const float*)d_in[28], (const float*)d_in[29], (float*)d_out);
}
